// sLSTMCell_35450660061940
// MI455X (gfx1250) — matrix operand order repaired and checked by reading, not compiled, not run on hardware
//
#include <hip/hip_runtime.h>
#include <hip/hip_bf16.h>

// ---------------------------------------------------------------------------
// sLSTM cell (B=4096, I=512, H=1024) for gfx1250.
//  Pass 1: convert comb=[x|h_prev] and the 4 gate weights to bf16 in d_ws
//          (one-time O(MK+NK) conversion; both operands then live in L2).
//  Pass 2: fused 4-gate GEMM via v_wmma_f32_16x16x32_bf16 with double-buffered
//          LDS tiles staged by global_load_async_to_lds_b128 (ASYNCcnt),
//          plus in-register sLSTM epilogue.
// d_ws usage: 4096*1536*2 + 4*1024*1536*2 bytes ~= 25.2 MB.
// ---------------------------------------------------------------------------

typedef __attribute__((ext_vector_type(16))) __bf16        v16bf;
typedef __attribute__((ext_vector_type(8)))  __bf16        v8bf;
typedef __attribute__((ext_vector_type(8)))  float         v8f;
typedef __attribute__((ext_vector_type(4)))  float         v4f;
typedef __attribute__((ext_vector_type(4)))  unsigned int  v4u;

constexpr int Bdim = 4096;
constexpr int Idim = 512;
constexpr int Hdim = 1024;
constexpr int Kdim = Idim + Hdim;   // 1536
constexpr int BM = 128;             // block M tile
constexpr int BN = 32;              // block N tile
constexpr int BK = 32;              // K step (matches 16x16x32 WMMA)
constexpr int NGATE = 4;

// round-to-nearest-even float -> bf16 bits
__device__ __forceinline__ unsigned short f2bf(float f) {
    unsigned int u = __float_as_uint(f);
    u += 0x7FFFu + ((u >> 16) & 1u);
    return (unsigned short)(u >> 16);
}

// pack 8 consecutive f32 into 8 bf16 (4 dwords)
__device__ __forceinline__ v4u pack8(const float* __restrict__ src) {
    v4f f0 = *(const v4f*)(src);
    v4f f1 = *(const v4f*)(src + 4);
    v4u o;
    o[0] = (unsigned)f2bf(f0[0]) | ((unsigned)f2bf(f0[1]) << 16);
    o[1] = (unsigned)f2bf(f0[2]) | ((unsigned)f2bf(f0[3]) << 16);
    o[2] = (unsigned)f2bf(f1[0]) | ((unsigned)f2bf(f1[1]) << 16);
    o[3] = (unsigned)f2bf(f1[2]) | ((unsigned)f2bf(f1[3]) << 16);
    return o;
}

// ---- pass 1a: comb = [x | h_prev]  ->  bf16 [B][K] ----
__global__ __launch_bounds__(256)
void conv_comb_kernel(const float* __restrict__ x, const float* __restrict__ h,
                      unsigned short* __restrict__ out) {
    const size_t t   = (size_t)blockIdx.x * blockDim.x + threadIdx.x;  // B*K/8 threads
    const size_t row = t / (Kdim / 8);
    const int    k   = (int)(t % (Kdim / 8)) * 8;    // multiple of 8; I=512 is too
    const float* src = (k < Idim) ? (x + row * Idim + k)
                                  : (h + row * Hdim + (k - Idim));
    *(v4u*)(out + row * Kdim + k) = pack8(src);
}

// ---- pass 1b: one gate weight matrix f32 [H][K] -> bf16 ----
__global__ __launch_bounds__(256)
void conv_w_kernel(const float* __restrict__ w, unsigned short* __restrict__ out) {
    const size_t t   = (size_t)blockIdx.x * blockDim.x + threadIdx.x;  // H*K/8 threads
    const size_t off = t * 8;
    *(v4u*)(out + off) = pack8(w + off);
}

// ---- pass 2: fused 4-gate WMMA GEMM + sLSTM epilogue ----
__global__ __launch_bounds__(256)
void slstm_gemm_kernel(const unsigned short* __restrict__ combBf,  // [B][K] bf16
                       const unsigned short* __restrict__ wBf,     // [4][H][K] bf16
                       const float* __restrict__ c_prev, const float* __restrict__ m_prev,
                       const float* __restrict__ n_prev,
                       const float* __restrict__ bi, const float* __restrict__ bfv,
                       const float* __restrict__ bz, const float* __restrict__ bo,
                       float* __restrict__ h_out, float* __restrict__ c_out,
                       float* __restrict__ m_out, float* __restrict__ n_out)
{
    // double-buffered bf16 tiles: A 2x(128x32)=16KB, B 2x(4x32x32)=16KB
    __shared__ __align__(16) unsigned short Alds[2][BM][BK];
    __shared__ __align__(16) unsigned short Blds[2][NGATE][BN][BK];

    const int tid  = threadIdx.x;
    const int lane = tid & 31;
    const int wave = tid >> 5;          // 0..7
    const int wm   = wave & 3;          // M sub-tile (32 rows)   -> 4*32 = 128
    const int wn   = wave >> 2;         // N sub-tile (16 cols)   -> 2*16 = 32
    const int half = lane >> 4;
    const int l16  = lane & 15;

    const int blockM = blockIdx.y * BM;
    const int blockN = blockIdx.x * BN;

    v8f acc[2][NGATE] = {};             // [m-subtile][gate i,f,z,o]

    // stage one K-tile into LDS buffer `buf` via async copy (ASYNCcnt).
    // 1024 b128 chunks per tile pair (A:512, B:512) / 256 threads = 4 asyncs/thread.
    auto stage = [&](int buf, int k0) {
        #pragma unroll
        for (int j = 0; j < 2; ++j) {
            const int c  = tid + 256 * j;          // 0..511
            const int ar = c >> 2, as = (c & 3) * 8;
            unsigned aoff = (unsigned)(uintptr_t)&Alds[buf][ar][as];
            const unsigned short* ga = combBf + (size_t)(blockM + ar) * Kdim + k0 + as;
            asm volatile("global_load_async_to_lds_b128 %0, %1, off"
                         :: "v"(aoff), "v"(ga) : "memory");
            const int bg = c >> 7, br = (c >> 2) & 31, bs = (c & 3) * 8;
            unsigned boff = (unsigned)(uintptr_t)&Blds[buf][bg][br][bs];
            const unsigned short* gb =
                wBf + ((size_t)bg * Hdim + blockN + br) * Kdim + k0 + bs;
            asm volatile("global_load_async_to_lds_b128 %0, %1, off"
                         :: "v"(boff), "v"(gb) : "memory");
        }
    };

    stage(0, 0);

    constexpr int KT = Kdim / BK;       // 48
    for (int kt = 0; kt < KT; ++kt) {
        const int buf = kt & 1;
        if (kt + 1 < KT) {
            stage(buf ^ 1, (kt + 1) * BK);
            // 4 newly-issued ops outstanding; older (current buf) must be done
            asm volatile("s_wait_asynccnt 0x4" ::: "memory");
        } else {
            asm volatile("s_wait_asynccnt 0x0" ::: "memory");
        }
        __syncthreads();                // all waves' current-buf data visible

        // A fragments (16x32 bf16): lane row M=l16; elems 0..7 at k=half*8,
        // elems 8..15 at k=half*8+16 (ISA 16-bit A layout)
        v16bf afrag[2];
        #pragma unroll
        for (int t = 0; t < 2; ++t) {
            const unsigned short* ap = &Alds[buf][wm * 32 + t * 16 + l16][half * 8];
            union { v16bf v; v8bf h[2]; } a;
            a.h[0] = *(const v8bf*)(ap);
            a.h[1] = *(const v8bf*)(ap + 16);
            afrag[t] = a.v;
        }
        // B fragments (32x16 bf16): lane col N=l16; elems 0..7 at k=half*8,
        // elems 8..15 at k=half*8+16 (same K order as A)
        #pragma unroll
        for (int g = 0; g < NGATE; ++g) {
            const unsigned short* bp = &Blds[buf][g][wn * 16 + l16][half * 8];
            union { v16bf v; v8bf h[2]; } b;
            b.h[0] = *(const v8bf*)(bp);
            b.h[1] = *(const v8bf*)(bp + 16);
            #pragma unroll
            for (int t = 0; t < 2; ++t) {
                acc[t][g] = __builtin_amdgcn_wmma_f32_16x16x32_bf16(
                    false, afrag[t], false, b.v, (short)0, acc[t][g], false, false);
            }
        }
        __syncthreads();                // reads done before next overwrite
    }

    // ---- fused sLSTM epilogue, all in registers ----
    const int col = blockN + wn * 16 + l16;
    const float vbi = bi[col],  vbf = bfv[col];
    const float vbz = bz[col],  vbo = bo[col];

    #pragma unroll
    for (int t = 0; t < 2; ++t) {
        #pragma unroll
        for (int r = 0; r < 8; ++r) {
            const int row = blockM + wm * 32 + t * 16 + half * 8 + r;  // C: M=r+8*half
            const size_t idx = (size_t)row * Hdim + col;

            const float log_i = acc[t][0][r] + vbi;
            const float log_f = acc[t][1][r] + vbf;
            const float mp    = m_prev[idx];
            const float m_t   = fmaxf(log_f + mp, log_i);
            const float i_t   = __expf(log_i - m_t);
            const float f_t   = __expf(log_f + mp - m_t);
            const float z_t   = tanhf(acc[t][2][r] + vbz);
            const float o_t   = 1.0f / (1.0f + __expf(-(acc[t][3][r] + vbo)));
            const float c_t   = f_t * c_prev[idx] + i_t * z_t;
            const float n_t   = f_t * n_prev[idx] + i_t;
            const float h_t   = o_t * (c_t / n_t);

            h_out[idx] = h_t;
            c_out[idx] = c_t;
            m_out[idx] = m_t;
            n_out[idx] = n_t;
        }
    }
}

extern "C" void kernel_launch(void* const* d_in, const int* in_sizes, int n_in,
                              void* d_out, int out_size, void* d_ws, size_t ws_size,
                              hipStream_t stream) {
    const float* x      = (const float*)d_in[0];
    const float* h_prev = (const float*)d_in[1];
    const float* c_prev = (const float*)d_in[2];
    const float* m_prev = (const float*)d_in[3];
    const float* n_prev = (const float*)d_in[4];
    const float* W[4]   = {(const float*)d_in[5], (const float*)d_in[7],
                           (const float*)d_in[9], (const float*)d_in[11]};
    const float* bi     = (const float*)d_in[6];
    const float* bfv    = (const float*)d_in[8];
    const float* bz     = (const float*)d_in[10];
    const float* bo     = (const float*)d_in[12];

    unsigned short* combBf = (unsigned short*)d_ws;                 // [B][K] bf16
    unsigned short* wBf    = combBf + (size_t)Bdim * Kdim;          // [4][H][K] bf16

    float* h_out = (float*)d_out;
    float* c_out = h_out + (size_t)Bdim * Hdim;
    float* m_out = c_out + (size_t)Bdim * Hdim;
    float* n_out = m_out + (size_t)Bdim * Hdim;

    // pass 1: one-time f32 -> bf16 conversions into workspace
    conv_comb_kernel<<<(Bdim * Kdim / 8) / 256, 256, 0, stream>>>(x, h_prev, combBf);
    for (int g = 0; g < 4; ++g) {
        conv_w_kernel<<<(Hdim * Kdim / 8) / 256, 256, 0, stream>>>(
            W[g], wBf + (size_t)g * Hdim * Kdim);
    }

    // pass 2: fused 4-gate WMMA GEMM + epilogue
    dim3 grid(Hdim / BN, Bdim / BM);   // (32, 32)
    slstm_gemm_kernel<<<grid, 256, 0, stream>>>(
        combBf, wBf, c_prev, m_prev, n_prev,
        bi, bfv, bz, bo, h_out, c_out, m_out, n_out);
}
